// GraphConvBlock_1211180777897
// MI455X (gfx1250) — matrix operand order repaired and checked by reading, not compiled, not run on hardware
//
#include <hip/hip_runtime.h>
#include <hip/hip_bf16.h>

typedef __attribute__((ext_vector_type(16))) _Float16 v16h;
typedef __attribute__((ext_vector_type(8)))  float    v8f;

#define Hdim 256

// ---------------- degree / scale ----------------

__global__ void degree_kernel(const int* __restrict__ row, float* __restrict__ deg, int E) {
    int e = blockIdx.x * blockDim.x + threadIdx.x;
    if (e < E) atomicAdd(&deg[row[e]], 1.0f);
}

__global__ void maxdeg_kernel(const float* __restrict__ deg, unsigned* __restrict__ maxbits, int n) {
    __shared__ float sm[256];
    int tid = blockIdx.x * blockDim.x + threadIdx.x;
    float v = (tid < n) ? deg[tid] : 0.0f;
    sm[threadIdx.x] = v;
    __syncthreads();
    for (int s = 128; s > 0; s >>= 1) {
        if (threadIdx.x < s) sm[threadIdx.x] = fmaxf(sm[threadIdx.x], sm[threadIdx.x + s]);
        __syncthreads();
    }
    if (threadIdx.x == 0) atomicMax(maxbits, __float_as_uint(sm[0]));  // deg >= 0 so bit-compare ok
}

__global__ void finalize_scale(const unsigned* __restrict__ maxbits, float* __restrict__ scale) {
    // lam = 2*max_deg ; scale = 2/lam = 1/max_deg
    *scale = 1.0f / __uint_as_float(*maxbits);
}

// ---------------- edge aggregation ----------------

// one block per edge, 256 lanes copy one contiguous 1KB feature row with f32 atomics
__global__ void aggregate256(const float* __restrict__ X, const int* __restrict__ row,
                             const int* __restrict__ col, float* __restrict__ agg) {
    int e = blockIdx.x;
    int c = threadIdx.x;
    int r = row[e], s = col[e];
    atomicAdd(&agg[(size_t)r * Hdim + c], X[(size_t)s * Hdim + c]);
}

__global__ void aggregate3(const float* __restrict__ X, const int* __restrict__ row,
                           const int* __restrict__ col, float* __restrict__ agg, int E) {
    int e = blockIdx.x * blockDim.x + threadIdx.x;
    if (e >= E) return;
    int r = row[e], s = col[e];
    atomicAdd(&agg[r * 3 + 0], X[s * 3 + 0]);
    atomicAdd(&agg[r * 3 + 1], X[s * 3 + 1]);
    atomicAdd(&agg[r * 3 + 2], X[s * 3 + 2]);
}

// tx1 = diag_coef[i]*x - scale*agg, computed in place over agg
__global__ void tx1_inplace(float* __restrict__ agg, const float* __restrict__ X,
                            const float* __restrict__ deg, const float* __restrict__ scale_p,
                            int total) {
    int tid = blockIdx.x * blockDim.x + threadIdx.x;
    if (tid >= total) return;
    float sc = *scale_p;
    int i = tid >> 8;                      // / Hdim
    float diag = sc * deg[i] - 1.0f;
    agg[tid] = diag * X[tid] - sc * agg[tid];
}

// ---------------- weight packing into WMMA B-fragment layout ----------------
// 12 matrices of HxH: mats 0..5 = Wr0[0..5], mats 6..11 = Wr1[0..5]
// packed[(((kt*16 + nt)*32 + lane)*16 + i)]  with
//   k = kt*32 + (i>=8 ? 16 : 0) + (lane>=16 ? 8 : 0) + (i&7) ,  n = nt*16 + (lane&15)
__global__ void pack_weights(const float* __restrict__ Wr0, const float* __restrict__ Wr1,
                             _Float16* __restrict__ out, int total) {
    int tid = blockIdx.x * blockDim.x + threadIdx.x;
    if (tid >= total) return;
    int i    = tid & 15;
    int lane = (tid >> 4) & 31;
    int tile = (tid >> 9) & 127;           // 8 ktiles * 16 ntiles
    int mat  = tid >> 16;                  // 65536 halfs per matrix
    int nt = tile & 15, kt = tile >> 4;
    int k = kt * 32 + ((i >> 3) * 16) + ((lane >> 4) * 8) + (i & 7);
    int n = nt * 16 + (lane & 15);
    const float* W = (mat < 6) ? (Wr0 + (size_t)mat * Hdim * Hdim)
                               : (Wr1 + (size_t)(mat - 6) * Hdim * Hdim);
    out[tid] = (_Float16)W[(size_t)k * Hdim + n];
}

// ---------------- WMMA cheb GEMM: dst = epilogue( X@W0 + T@W1 + bias ) ----------------
// block = (32,4): 4 waves, each wave owns a 16x64 strip of the 16x256 block output.
// mode 0: dst = relu(v+b)        mode 1: dst = (dst + relu(v+b)) * 0.5   (residual into y)
__global__ __launch_bounds__(128) void cheb_gemm_wmma(
    const float* __restrict__ X, const float* __restrict__ T,
    const _Float16* __restrict__ W0p, const _Float16* __restrict__ W1p,
    const float* __restrict__ bias, float* __restrict__ dst, int mode, int Nrows) {
    const int lane = threadIdx.x;
    const int wave = threadIdx.y;
    const int m0 = blockIdx.x * 16;
    if (m0 >= Nrows) return;
    const int n0 = wave * 64;
    const int mrow = lane & 15;
    const int hi = lane >> 4;              // 0 or 1

    v8f acc[4];
#pragma unroll
    for (int t = 0; t < 4; ++t) acc[t] = (v8f){0.f,0.f,0.f,0.f,0.f,0.f,0.f,0.f};

#pragma unroll
    for (int half = 0; half < 2; ++half) {
        const float* src = half ? T : X;
        const _Float16* wp = half ? W1p : W0p;
        const float* arow = src + (size_t)(m0 + mrow) * Hdim + hi * 8;
#pragma unroll
        for (int kt = 0; kt < 8; ++kt) {
            const float* ap = arow + kt * 32;
            v16h a;
#pragma unroll
            for (int j = 0; j < 8; ++j) a[j] = (_Float16)ap[j];        // K = kt*32 + hi*8 + j
#pragma unroll
            for (int j = 0; j < 8; ++j) a[8 + j] = (_Float16)ap[16 + j]; // K = kt*32 + 16 + hi*8 + j
            const _Float16* wbase = wp + (((size_t)kt * 16 + (n0 >> 4)) * 32 + lane) * 16;
#pragma unroll
            for (int t = 0; t < 4; ++t) {
                v16h b = *(const v16h*)(wbase + (size_t)t * 512);
                acc[t] = __builtin_amdgcn_wmma_f32_16x16x32_f16(
                    false, a, false, b, (short)0, acc[t], false, false);
            }
        }
    }

    const int mbase = m0 + (hi ? 8 : 0);
#pragma unroll
    for (int t = 0; t < 4; ++t) {
        const int n = n0 + t * 16 + (lane & 15);
        const float bv = bias[n];
#pragma unroll
        for (int r = 0; r < 8; ++r) {
            float v = acc[t][r] + bv;
            v = v > 0.0f ? v : 0.0f;
            size_t o = (size_t)(mbase + r) * Hdim + n;
            if (mode) dst[o] = (dst[o] + v) * 0.5f;
            else      dst[o] = v;
        }
    }
}

// ---------------- small input (K=3) and output (->3) cheb layers ----------------

__global__ void cheb_input(const float* __restrict__ x, const float* __restrict__ agg3,
                           const float* __restrict__ deg, const float* __restrict__ scale_p,
                           const float* __restrict__ Wi0, const float* __restrict__ Wi1,
                           const float* __restrict__ bi, float* __restrict__ y, int total) {
    int tid = blockIdx.x * blockDim.x + threadIdx.x;
    if (tid >= total) return;
    int n = tid & 255;
    int i = tid >> 8;
    float sc = *scale_p;
    float diag = sc * deg[i] - 1.0f;
    float acc = bi[n];
#pragma unroll
    for (int k = 0; k < 3; ++k) {
        float xv = x[i * 3 + k];
        float t1 = diag * xv - sc * agg3[i * 3 + k];
        acc += xv * Wi0[k * Hdim + n] + t1 * Wi1[k * Hdim + n];
    }
    y[tid] = acc > 0.0f ? acc : 0.0f;
}

__global__ void cheb_final(const float* __restrict__ y, const float* __restrict__ T,
                           const float* __restrict__ Wf0, const float* __restrict__ Wf1,
                           const float* __restrict__ bf, float* __restrict__ y2, int total) {
    int tid = blockIdx.x * blockDim.x + threadIdx.x;
    if (tid >= total) return;
    int c = tid % 3;
    int i = tid / 3;
    const float* yr = y + (size_t)i * Hdim;
    const float* tr = T + (size_t)i * Hdim;
    float acc = bf[c];
    for (int k = 0; k < Hdim; ++k)
        acc += yr[k] * Wf0[k * 3 + c] + tr[k] * Wf1[k * 3 + c];
    y2[tid] = acc;
}

// ---------------- launch ----------------

extern "C" void kernel_launch(void* const* d_in, const int* in_sizes, int n_in,
                              void* d_out, int out_size, void* d_ws, size_t ws_size,
                              hipStream_t stream) {
    const int N = in_sizes[0] / 3;     // 50000
    const int E = in_sizes[1] / 2;     // 800000

    const float* x   = (const float*)d_in[0];
    const int*   ei  = (const int*)  d_in[1];
    const int*   row = ei;
    const int*   col = ei + E;
    const float* Wi0 = (const float*)d_in[2];
    const float* Wi1 = (const float*)d_in[3];
    const float* bi  = (const float*)d_in[4];
    const float* Wr0 = (const float*)d_in[5];
    const float* Wr1 = (const float*)d_in[6];
    const float* br  = (const float*)d_in[7];
    const float* Wf0 = (const float*)d_in[8];
    const float* Wf1 = (const float*)d_in[9];
    const float* bf  = (const float*)d_in[10];

    float* out = (float*)d_out;
    float* y2  = out;                   // [N,3]
    float* y   = out + (size_t)N * 3;   // [N,256] -- second reference output, used as live buffer

    // workspace carve-out (256B aligned regions)
    char*  ws  = (char*)d_ws;
    size_t off = 0;
    auto carve = [&](size_t bytes) { char* p = ws + off; off = (off + bytes + 255) & ~(size_t)255; return p; };
    float*     deg     = (float*)    carve((size_t)N * 4);
    unsigned*  maxbits = (unsigned*) carve(256);           // [0]=max bits, [1]=scale
    float*     scale   = (float*)(maxbits + 1);
    float*     agg     = (float*)    carve((size_t)N * Hdim * 4);
    float*     h       = (float*)    carve((size_t)N * Hdim * 4);
    _Float16*  packW   = (_Float16*) carve((size_t)12 * Hdim * Hdim * 2);

    const int NH = N * Hdim;

    // degrees + lambda_max -> scale
    hipMemsetAsync(deg, 0, (size_t)N * 4, stream);
    hipMemsetAsync(maxbits, 0, 8, stream);
    degree_kernel<<<(E + 255) / 256, 256, 0, stream>>>(row, deg, E);
    maxdeg_kernel<<<(N + 255) / 256, 256, 0, stream>>>(deg, maxbits, N);
    finalize_scale<<<1, 1, 0, stream>>>(maxbits, scale);

    // pack all recurrent weights into WMMA B-fragment layout (f16)
    pack_weights<<<(12 * 65536 + 255) / 256, 256, 0, stream>>>(Wr0, Wr1, packW, 12 * 65536);

    // input cheb: x[N,3] -> y[N,256]
    hipMemsetAsync(agg, 0, (size_t)N * 3 * 4, stream);
    aggregate3<<<(E + 255) / 256, 256, 0, stream>>>(x, row, col, agg, E);
    cheb_input<<<(NH + 255) / 256, 256, 0, stream>>>(x, agg, deg, scale, Wi0, Wi1, bi, y, NH);

    dim3 gblk(32, 4, 1);
    int  ggrid = (N + 15) / 16;

    for (int i = 0; i < 3; ++i) {
        // h = relu(cheb(y))
        hipMemsetAsync(agg, 0, (size_t)NH * 4, stream);
        aggregate256<<<E, 256, 0, stream>>>(y, row, col, agg);
        tx1_inplace<<<(NH + 255) / 256, 256, 0, stream>>>(agg, y, deg, scale, NH);
        cheb_gemm_wmma<<<ggrid, gblk, 0, stream>>>(
            y, agg, packW + (size_t)(2 * i) * 65536, packW + (size_t)(6 + 2 * i) * 65536,
            br + (size_t)(2 * i) * Hdim, h, 0, N);
        // y = (y + relu(cheb(h))) / 2
        hipMemsetAsync(agg, 0, (size_t)NH * 4, stream);
        aggregate256<<<E, 256, 0, stream>>>(h, row, col, agg);
        tx1_inplace<<<(NH + 255) / 256, 256, 0, stream>>>(agg, h, deg, scale, NH);
        cheb_gemm_wmma<<<ggrid, gblk, 0, stream>>>(
            h, agg, packW + (size_t)(2 * i + 1) * 65536, packW + (size_t)(6 + 2 * i + 1) * 65536,
            br + (size_t)(2 * i + 1) * Hdim, y, 1, N);
    }

    // final cheb: y -> y2[N,3] (no relu)
    hipMemsetAsync(agg, 0, (size_t)NH * 4, stream);
    aggregate256<<<E, 256, 0, stream>>>(y, row, col, agg);
    tx1_inplace<<<(NH + 255) / 256, 256, 0, stream>>>(agg, y, deg, scale, NH);
    cheb_final<<<(N * 3 + 255) / 256, 256, 0, stream>>>(y, agg, Wf0, Wf1, bf, y2, N * 3);
}
